// MeshMultiHeadHodgeAttentionVertices_6408091206162
// MI455X (gfx1250) — matrix operand order repaired and checked by reading, not compiled, not run on hardware
//
#include <hip/hip_runtime.h>

// ---------------------------------------------------------------------------
// CDNA5 (gfx1250, wave32) MeshMultiHeadHodgeAttentionVertices
// B=2, N=4096, M=12288, H=8, D=256, DK=32, K_nbr=16
//
// - All GEMMs: v_wmma_f32_16x16x32_bf16 (f32 accumulate), A pre-converted to
//   bf16 so the A tile is a raw byte copy -> GLOBAL_LOAD_ASYNC_TO_LDS_B128,
//   double-buffered, stored k-contiguous (fragment-contiguous) in LDS.
// - d_0 converted once into bf16 (M,N) and bf16-transposed (N,M): each batch
//   slice (100 MB) fits the 192 MB L2, and both big GEMMs use one code path.
// - LayerNorm(32) / 16-neighbor sparse attention: one wave32 per head vector.
// ---------------------------------------------------------------------------

typedef __attribute__((ext_vector_type(16))) __bf16 v16bf;
typedef __attribute__((ext_vector_type(8)))  float  v8f;
typedef __attribute__((ext_vector_type(4)))  int    vi4;   // async-copy unit

union FragB16 {
  v16bf v;
  uint4 q[2];
};

#if defined(__has_builtin)
#  if __has_builtin(__builtin_amdgcn_global_load_async_to_lds_b128)
#    define HAVE_ASYNC_LDS 1
#  endif
#  if __has_builtin(__builtin_amdgcn_s_wait_asynccnt)
#    define HAVE_WAIT_ASYNC 1
#  endif
#endif

static __device__ __forceinline__ void async_wait_all()
{
#if defined(HAVE_ASYNC_LDS)
#  if defined(HAVE_WAIT_ASYNC)
  __builtin_amdgcn_s_wait_asynccnt(0);
#  else
  asm volatile("s_wait_asynccnt 0x0" ::: "memory");
#  endif
#endif
}

static __device__ __forceinline__ unsigned pk_bf16(float a, float b)
{
  union { __bf16 h[2]; unsigned u; } x;
  x.h[0] = (__bf16)a; x.h[1] = (__bf16)b;
  return x.u;
}

#define ASTRIDE 40   // bf16 elems per LDS row: 80 B, 16B-aligned, conflict-free

// ---------------------------------------------------------------------------
// GEMM: C[rows x 256] = A(bf16) @ B'(fp32->bf16), B' = TB ? B^T : B.
// Block tile 128x128, K-tile 32, 256 threads = 8 waves (grid of 4x2 wave
// tiles, each 32 rows x 64 cols = 2 A-frags x 4 B-frags -> 8 WMMAs).
//
// LDS A row layout is in plain k order so a lane's 16-bf16 fragment is
// contiguous:   bytes [0,16)=k0-7  [16,32)=k8-15  [32,48)=k16-23  [48,64)=k24-31
// (k-octet o lands in slot o); fragment for lane-half lh reads
// elements [16*lh, 16*lh+16).  B is [col][k], k contiguous, same property.
// ---------------------------------------------------------------------------
template <bool TB>
__global__ __launch_bounds__(256)
void gemm_abf16_wmma(const __bf16* __restrict__ A, long sAb, int lda,
                     const float* __restrict__ B, long sBb, int ldb,
                     float* __restrict__ C, long sCb, int ldc,
                     int K)
{
  __shared__ __align__(16) __bf16 As[2][128 * ASTRIDE];
  __shared__ __align__(16) __bf16 Bs[2][128 * ASTRIDE];

  const int tid  = threadIdx.x;
  const int lane = tid & 31;
  const int wave = tid >> 5;
  const int lr   = lane & 15;
  const int lh   = lane >> 4;
  const int rg   = wave >> 1;          // 0..3 : 32-row group
  const int cg   = wave & 1;           // 0..1 : 64-col group

  const int rowBase = blockIdx.y * 128;
  const int colBase = blockIdx.x * 128;
  A += (size_t)blockIdx.z * sAb;
  B += (size_t)blockIdx.z * sBb;
  C += (size_t)blockIdx.z * sCb;

  // ---- A tile stage: 128 rows x 32 k (bf16) = 512 x b128 async copies ----
  auto stage_A = [&](int k0, int bufi) {
#pragma unroll
    for (int half = 0; half < 2; ++half) {
      const int i    = tid + half * 256;
      const int row  = i >> 2;
      const int o    = i & 3;                       // k-octet
      const int slot = o;                           // contiguous
      const __bf16* g = A + (size_t)(rowBase + row) * lda + (k0 + 8 * o);
      __bf16*       l = &As[bufi][row * ASTRIDE + 8 * slot];
#if defined(HAVE_ASYNC_LDS)
      __builtin_amdgcn_global_load_async_to_lds_b128((vi4*)g, (vi4*)l, 0, 0);
#else
      *reinterpret_cast<uint4*>(l) = *reinterpret_cast<const uint4*>(g);
#endif
    }
  };

  // ---- B tile stage: fp32 -> bf16, LDS [col][k] ----
  auto stage_B = [&](int k0, int bufi) {
    if (!TB) {                       // B is (K x 256) row-major
      const int c  = tid & 127;
      const int kb = tid >> 7;       // 0/1
#pragma unroll
      for (int rep = 0; rep < 4; ++rep) {
        const int k4 = (kb + 2 * rep) * 4;          // covers k 0..31 by quads
        const float* s = B + (size_t)(k0 + k4) * ldb + colBase + c;
        float f0 = s[0], f1 = s[ldb], f2 = s[2 * ldb], f3 = s[3 * (size_t)ldb];
        uint2 u; u.x = pk_bf16(f0, f1); u.y = pk_bf16(f2, f3);
        *reinterpret_cast<uint2*>(&Bs[bufi][c * ASTRIDE + k4]) = u;
      }
    } else {                         // B is (256 x K) row-major (weights W)
      const int c  = tid >> 1;
      const int kq = (tid & 1) << 4;
      const float* s = B + (size_t)(colBase + c) * ldb + (k0 + kq);
#pragma unroll
      for (int i = 0; i < 4; ++i) {
        float4 f = *reinterpret_cast<const float4*>(s + 4 * i);
        uint2 u; u.x = pk_bf16(f.x, f.y); u.y = pk_bf16(f.z, f.w);
        *reinterpret_cast<uint2*>(&Bs[bufi][c * ASTRIDE + kq + 4 * i]) = u;
      }
    }
  };

  v8f acc[2][4] = {};

  // prologue: stage tile 0
  stage_B(0, 0);
  stage_A(0, 0);
  async_wait_all();
  __syncthreads();

  int buf = 0;
  for (int k0 = 0; k0 < K; k0 += 32) {
    const int nxt = buf ^ 1;
    if (k0 + 32 < K) {               // stage next tile while computing
      stage_B(k0 + 32, nxt);
      stage_A(k0 + 32, nxt);
    }

    // ---- fragments: contiguous 32B per lane -> 2x ds_load_b128 each ----
    FragB16 a[2], bb[4];
#pragma unroll
    for (int ii = 0; ii < 2; ++ii) {
      const __bf16* p = &As[buf][(rg * 32 + ii * 16 + lr) * ASTRIDE + 16 * lh];
      a[ii].q[0] = *reinterpret_cast<const uint4*>(p);
      a[ii].q[1] = *reinterpret_cast<const uint4*>(p + 8);
    }
#pragma unroll
    for (int t = 0; t < 4; ++t) {
      const __bf16* p = &Bs[buf][(cg * 64 + t * 16 + lr) * ASTRIDE + 16 * lh];
      bb[t].q[0] = *reinterpret_cast<const uint4*>(p);
      bb[t].q[1] = *reinterpret_cast<const uint4*>(p + 8);
    }
#pragma unroll
    for (int ii = 0; ii < 2; ++ii)
#pragma unroll
      for (int t = 0; t < 4; ++t)
        acc[ii][t] = __builtin_amdgcn_wmma_f32_16x16x32_bf16(
            false, a[ii].v, false, bb[t].v, (short)0, acc[ii][t], false, false);

    async_wait_all();
    __syncthreads();
    buf = nxt;
  }

  // ---- epilogue: C/D layout vgpr v -> row v + 8*lh, col = lane&15 ----
#pragma unroll
  for (int ii = 0; ii < 2; ++ii)
#pragma unroll
    for (int t = 0; t < 4; ++t)
#pragma unroll
      for (int v = 0; v < 8; ++v) {
        const int r = rowBase + rg * 32 + ii * 16 + v + (lh << 3);
        const int c = colBase + cg * 64 + t * 16 + lr;
        C[(size_t)r * ldc + c] = acc[ii][t][v];
      }
}

// ---------------------------------------------------------------------------
// fp32 -> bf16 elementwise (4 elems / thread)
// ---------------------------------------------------------------------------
__global__ __launch_bounds__(256)
void cvt_bf16_kernel(const float* __restrict__ src, __bf16* __restrict__ dst,
                     long n4)
{
  const long i = (long)blockIdx.x * blockDim.x + threadIdx.x;
  if (i >= n4) return;
  float4 f = reinterpret_cast<const float4*>(src)[i];
  uint2 u; u.x = pk_bf16(f.x, f.y); u.y = pk_bf16(f.z, f.w);
  reinterpret_cast<uint2*>(dst)[i] = u;
}

// ---------------------------------------------------------------------------
// d_0 (B,M,N) fp32 -> dn (B,M,N) bf16  and  dt (B,N,M) bf16 (transposed),
// one read of d_0.  64x64 tiles via LDS.  grid (N/64, M/64, B), 256 thr.
// ---------------------------------------------------------------------------
__global__ __launch_bounds__(256)
void cvt_transpose_d0_kernel(const float* __restrict__ d0,
                             __bf16* __restrict__ dn,
                             __bf16* __restrict__ dt,
                             int Mdim, int Ndim)
{
  __shared__ float tile[64][65];
  const int tid = threadIdx.x;
  const int b   = blockIdx.z;
  const size_t base = (size_t)b * Mdim * Ndim;
  const int n0 = blockIdx.x * 64;
  const int m0 = blockIdx.y * 64;

  const int r  = tid >> 2;          // 0..63
  const int cq = (tid & 3) << 4;    // 0,16,32,48
#pragma unroll
  for (int i = 0; i < 4; ++i) {
    const int c = cq + 4 * i;
    float4 f = *reinterpret_cast<const float4*>(
        d0 + base + (size_t)(m0 + r) * Ndim + (n0 + c));
    tile[r][c + 0] = f.x; tile[r][c + 1] = f.y;
    tile[r][c + 2] = f.z; tile[r][c + 3] = f.w;
    uint2 u; u.x = pk_bf16(f.x, f.y); u.y = pk_bf16(f.z, f.w);
    *reinterpret_cast<uint2*>(dn + base + (size_t)(m0 + r) * Ndim + (n0 + c)) = u;
  }
  __syncthreads();
#pragma unroll
  for (int i = 0; i < 4; ++i) {
    const int c = cq + 4 * i;       // m offset inside tile
    float f0 = tile[c + 0][r], f1 = tile[c + 1][r];
    float f2 = tile[c + 2][r], f3 = tile[c + 3][r];
    uint2 u; u.x = pk_bf16(f0, f1); u.y = pk_bf16(f2, f3);
    *reinterpret_cast<uint2*>(
        dt + (size_t)b * Ndim * Mdim + (size_t)(n0 + r) * Mdim + (m0 + c)) = u;
  }
}

// ---------------------------------------------------------------------------
// Per-head LayerNorm: one wave32 per 32-element head vector (DK == warpSize).
// ---------------------------------------------------------------------------
__global__ __launch_bounds__(256)
void layernorm32_kernel(float* __restrict__ buf, int chunks)
{
  const int w    = (int)((blockIdx.x * blockDim.x + threadIdx.x) >> 5);
  const int lane = threadIdx.x & 31;
  if (w >= chunks) return;

  const size_t off = (size_t)w * 32 + lane;
  float v = buf[off];
  float s = v;
#pragma unroll
  for (int o = 16; o > 0; o >>= 1) s += __shfl_xor(s, o, 32);
  const float mu = s * (1.0f / 32.0f);
  const float d  = v - mu;
  float q = d * d;
#pragma unroll
  for (int o = 16; o > 0; o >>= 1) q += __shfl_xor(q, o, 32);
  buf[off] = d * __frsqrt_rn(q * (1.0f / 32.0f) + 1e-5f);
}

// ---------------------------------------------------------------------------
// 16-neighbor sparse attention: one wave32 per (b, pos, h).
// ---------------------------------------------------------------------------
__global__ __launch_bounds__(256)
void sparse_attn_kernel(const float* __restrict__ x,
                        const float* __restrict__ q,
                        const float* __restrict__ k,
                        const int*   __restrict__ idx,
                        float* __restrict__ out,
                        int seq, int total)
{
  const int w    = (int)((blockIdx.x * blockDim.x + threadIdx.x) >> 5);
  const int lane = threadIdx.x & 31;
  if (w >= total) return;

  const int h   = w & 7;
  const int pos = (w >> 3) % seq;
  const int b   = (w >> 3) / seq;
  const int col = h * 32 + lane;

  const size_t rowbase = ((size_t)b * seq + pos) * 256;
  const float  qv = q[rowbase + col];
  const int*   ip = idx + (size_t)pos * 16;

  float sc[16];
#pragma unroll
  for (int j = 0; j < 16; ++j) {
    float p = qv * k[((size_t)b * seq + ip[j]) * 256 + col];
#pragma unroll
    for (int o = 16; o > 0; o >>= 1) p += __shfl_xor(p, o, 32);
    sc[j] = p * 0.17677669529663687f;               // 1/sqrt(32)
  }
  float m = sc[0];
#pragma unroll
  for (int j = 1; j < 16; ++j) m = fmaxf(m, sc[j]);
  float se = 0.0f;
#pragma unroll
  for (int j = 0; j < 16; ++j) { sc[j] = __expf(sc[j] - m); se += sc[j]; }
  const float inv = 1.0f / se;

  float acc = 0.0f;
#pragma unroll
  for (int j = 0; j < 16; ++j)
    acc += (sc[j] * inv) * x[((size_t)b * seq + ip[j]) * 256 + col];
  out[rowbase + col] = acc;
}

// ---------------------------------------------------------------------------
// Host-side orchestration
// ---------------------------------------------------------------------------
extern "C" void kernel_launch(void* const* d_in, const int* in_sizes, int n_in,
                              void* d_out, int out_size, void* d_ws, size_t ws_size,
                              hipStream_t stream)
{
  (void)in_sizes; (void)n_in; (void)out_size; (void)ws_size;

  constexpr int B = 2, N = 4096, M = 12288, H = 8, D = 256;

  const float* x_v = (const float*)d_in[0];
  const float* x_e = (const float*)d_in[1];
  const float* d0  = (const float*)d_in[2];
  const float* WvQ = (const float*)d_in[3];
  const float* WvK = (const float*)d_in[4];
  const float* WvV = (const float*)d_in[5];
  const float* WeQ = (const float*)d_in[6];
  const float* WeK = (const float*)d_in[7];
  const int*   v_idx = (const int*)d_in[8];
  const int*   e_idx = (const int*)d_in[9];
  float* out = (float*)d_out;

  // ---- workspace layout ----
  const size_t szV = (size_t)B * N * D;   // 2,097,152
  const size_t szE = (size_t)B * M * D;   // 6,291,456
  const size_t szD = (size_t)B * M * N;   // 100,663,296
  float* wsf = (float*)d_ws;
  float* vQ  = wsf;
  float* vK  = vQ + szV;
  float* vV  = vK + szV;
  float* eQ  = vV + szV;
  float* eK  = eQ + szE;
  float* y1  = eK + szE;
  float* y1s = y1 + szE;
  float* y2  = y1;                               // y1 dead after e-attention
  __bf16* xvbf = (__bf16*)(y1s + szE);
  __bf16* xebf = xvbf + szV;
  __bf16* d0bf = xebf + szE;                     // (B, M, N) bf16
  __bf16* d0bt = d0bf + szD;                     // (B, N, M) bf16

  const dim3 blk(256);

  // ---- pre-convert activations & d_0 (both layouts) to bf16 ----
  cvt_bf16_kernel<<<(int)(szV / 4 / 256), blk, 0, stream>>>(x_v, xvbf, szV / 4);
  cvt_bf16_kernel<<<(int)(szE / 4 / 256), blk, 0, stream>>>(x_e, xebf, szE / 4);
  {
    dim3 g(N / 64, M / 64, B);
    cvt_transpose_d0_kernel<<<g, blk, 0, stream>>>(d0, d0bf, d0bt, M, N);
  }

  // ---- projections: (rows x 256) @ W^T  (einsum bnd,ed->bne => TB=true) ----
  {
    dim3 g(D / 128, (B * N) / 128, 1);
    gemm_abf16_wmma<true><<<g, blk, 0, stream>>>(xvbf, 0L, D, WvQ, 0L, D, vQ, 0L, D, D);
    gemm_abf16_wmma<true><<<g, blk, 0, stream>>>(xvbf, 0L, D, WvK, 0L, D, vK, 0L, D, D);
    gemm_abf16_wmma<true><<<g, blk, 0, stream>>>(xvbf, 0L, D, WvV, 0L, D, vV, 0L, D, D);
  }
  {
    dim3 g(D / 128, (B * M) / 128, 1);
    gemm_abf16_wmma<true><<<g, blk, 0, stream>>>(xebf, 0L, D, WeQ, 0L, D, eQ, 0L, D, D);
    gemm_abf16_wmma<true><<<g, blk, 0, stream>>>(xebf, 0L, D, WeK, 0L, D, eK, 0L, D, D);
  }
  // ---- per-head LayerNorm on vQ, vK, eQ, eK ----
  {
    const int cv = B * N * H, bv = (cv * 32) / 256;
    layernorm32_kernel<<<bv, blk, 0, stream>>>(vQ, cv);
    layernorm32_kernel<<<bv, blk, 0, stream>>>(vK, cv);
    const int ce = B * M * H, be = (ce * 32) / 256;
    layernorm32_kernel<<<be, blk, 0, stream>>>(eQ, ce);
    layernorm32_kernel<<<be, blk, 0, stream>>>(eK, ce);
  }
  // ---- y1[b] = d0[b] (MxN) @ vV[b] (Nx256) ----
  {
    dim3 g(D / 128, M / 128, B);
    gemm_abf16_wmma<false><<<g, blk, 0, stream>>>(
        d0bf, (long)M * N, N, vV, (long)N * D, D, y1, (long)M * D, D, N);
  }
  // ---- edge sparse attention: q=eQ, k=eK, x=y1 -> y1s ----
  {
    const int total = B * M * H;
    sparse_attn_kernel<<<total / 8, blk, 0, stream>>>(y1, eQ, eK, e_idx, y1s, M, total);
  }
  // ---- y2[b] = d0[b]^T (NxM) @ y1s[b] (Mx256) ----
  {
    dim3 g(D / 128, N / 128, B);
    gemm_abf16_wmma<false><<<g, blk, 0, stream>>>(
        d0bt, (long)N * M, M, y1s, (long)M * D, D, y2, (long)N * D, D, M);
  }
  // ---- vertex sparse attention: q=vK, k=vQ, x=y2 -> out ----
  {
    const int total = B * N * H;
    sparse_attn_kernel<<<total / 8, blk, 0, stream>>>(y2, vK, vQ, v_idx, out, N, total);
  }
}
